// MultiHeadAttentionLayerGritRoPE_62955630625330
// MI455X (gfx1250) — matrix operand order repaired and checked by reading, not compiled, not run on hardware
//
#include <hip/hip_runtime.h>
#include <math.h>

// ---------------- problem constants (from reference) ----------------
#define NN    50000     // nodes
#define NEDGE 500000    // edges
#define INF   128       // input feature dim
#define HH    8         // heads
#define DD    16        // head dim
#define KS    21        // rrwp dim
#define CLAMPV 5.0f

#define NODE_TILES (NN / 16)      // 3125, exact
#define EDGE_TILES (NEDGE / 16)   // 31250, exact

typedef _Float16 v16h __attribute__((ext_vector_type(16)));
typedef _Float16 v8h  __attribute__((ext_vector_type(8)));
typedef float    v8f  __attribute__((ext_vector_type(8)));

// ---------------- WMMA helpers ----------------
__device__ __forceinline__ v8f wmma_f16(v16h a, v16h b, v8f c) {
  // D = A(16x32 f16) x B(32x16 f16) + C(16x16 f32)
  return __builtin_amdgcn_wmma_f32_16x16x32_f16(
      /*neg_a=*/false, a, /*neg_b=*/false, b,
      /*c_mod=*/(short)0, c, /*reuse_a=*/false, /*reuse_b=*/false);
}

__device__ __forceinline__ v8h cvt8(float4 a, float4 b) {
  v8h h;
  h[0]=(_Float16)a.x; h[1]=(_Float16)a.y; h[2]=(_Float16)a.z; h[3]=(_Float16)a.w;
  h[4]=(_Float16)b.x; h[5]=(_Float16)b.y; h[6]=(_Float16)b.z; h[7]=(_Float16)b.w;
  return h;
}

__device__ __forceinline__ v16h cat16(v8h lo, v8h hi) {
  return __builtin_shufflevector(lo, hi, 0,1,2,3,4,5,6,7,8,9,10,11,12,13,14,15);
}

// A-fragment (16x32 f16): lane l holds row M=l&15; with base=8*(l>>4),
// elements 0..7 = K base..base+7, elements 8..15 = K 16+base..16+base+7.
// rowp = pre-converted f16 row; k = kstep*32 + base.
__device__ __forceinline__ v16h loadA_f16row(const _Float16* __restrict__ rowp, int k) {
  v8h lo = *(const v8h*)(rowp + k);
  v8h hi = *(const v8h*)(rowp + k + 16);
  return cat16(lo, hi);
}

// B-fragment (32x16 f16): lane l holds column N=l&15; with base=8*(l>>4),
// elements 0..7 = K base..base+7, elements 8..15 = K 16+base..16+base+7.
__device__ __forceinline__ v16h loadB_lds(const _Float16* base) {
  v8h lo = *(const v8h*)(base);
  v8h hi = *(const v8h*)(base + 16);
  return cat16(lo, hi);
}

__device__ __forceinline__ float vsel8(v8f v, int i) {
  float r = v[0];
#pragma unroll
  for (int j = 1; j < 8; ++j) r = (i == j) ? v[j] : r;
  return r;
}

// ---------------- kernel 0: zero den + wV ----------------
__global__ void grit_zero_kernel(float* wV, float* den) {
  size_t i = (size_t)blockIdx.x * blockDim.x + threadIdx.x;
  if (i < (size_t)NN * 128) wV[i] = 0.0f;
  if (i < (size_t)NN * HH)  den[i] = 0.0f;
}

// ---------------- kernel 0b: f32 -> f16 weight conversion (one-shot) ----
__global__ void grit_cvt_kernel(const float* __restrict__ src,
                                _Float16* __restrict__ dst, int n) {
  int i = blockIdx.x * blockDim.x + threadIdx.x;
  if (i < n) dst[i] = (_Float16)src[i];
}

// ---------------- kernel 1: node QKV projections + RoPE ----------------
// grid = NODE_TILES blocks x 256 threads (8 waves; wave h handles head h).
// C tile is transposed: M = output feature row, N = node.
__global__ void grit_node_kernel(const float* __restrict__ x,
                                 const float* __restrict__ rrwp,
                                 const _Float16* __restrict__ Qh,
                                 const _Float16* __restrict__ Kh,
                                 const _Float16* __restrict__ Vh,
                                 const float* __restrict__ Qb, const float* __restrict__ Kb,
                                 const float* __restrict__ Vb,
                                 const float* __restrict__ Wang,
                                 float* __restrict__ Qws, float* __restrict__ Kws,
                                 float* __restrict__ Vws) {
  __shared__ _Float16 xt[16 * 128];   // x tile, f16, row-major
  const int n0 = blockIdx.x * 16;
  const int t  = threadIdx.x;

  // stage x tile -> LDS (each thread converts 8 floats)
  {
    int idx = t * 8;
    int row = idx >> 7;
    int colc = idx & 127;
    const float4* p = (const float4*)(x + (size_t)(n0 + row) * INF + colc);
    float4 a = p[0], b = p[1];
    *(v8h*)&xt[idx] = cvt8(a, b);
  }
  __syncthreads();

  const int lane = t & 31;
  const int h    = t >> 5;        // head / wave id
  const int col  = lane & 15;     // node within tile (C column)
  const int half = lane >> 4;     // M-half of C tile this lane holds
  const int node = n0 + col;

  v8f cq = {0,0,0,0,0,0,0,0};
  v8f ck = {0,0,0,0,0,0,0,0};
  v8f cv = {0,0,0,0,0,0,0,0};

  const _Float16* qrow = Qh + (size_t)(h * 16 + col) * INF;
  const _Float16* krow = Kh + (size_t)(h * 16 + col) * INF;
  const _Float16* vrow = Vh + (size_t)(h * 16 + col) * INF;

#pragma unroll 1
  for (int kk = 0; kk < 4; ++kk) {
    v16h B = loadB_lds(&xt[col * 128 + kk * 32 + half * 8]);
    int ka = kk * 32 + half * 8;
    cq = wmma_f16(loadA_f16row(qrow, ka), B, cq);
    ck = wmma_f16(loadA_f16row(krow, ka), B, ck);
    cv = wmma_f16(loadA_f16row(vrow, ka), B, cv);
  }

  // biases (zero-valued in reference, kept for fidelity)
#pragma unroll
  for (int r = 0; r < 8; ++r) {
    int m = h * 16 + half * 8 + r;
    cq[r] += Qb[m];
    ck[r] += Kb[m];
    cv[r] += Vb[m];
  }

  // RoPE angles for this (node, head): angles[p] = sum_k rrwp[n,k]*Wang[h,k,p]
  // lane-half 0 holds dims 0..7  -> rotation pairs p=0,1 on (m=4,5),(6,7)
  // lane-half 1 holds dims 8..15 -> pairs p=2..5 on (8,9),(10,11),(12,13),(14,15)
  // Rolled loop with scalar loads: keeps only ang[4] + pointers live (no spills).
  float ang[4] = {0.f, 0.f, 0.f, 0.f};
  {
    const float* wa = Wang + (size_t)h * KS * 6 + (half ? 2 : 0);
    const float* rr = rrwp + (size_t)node * KS;
#pragma unroll 1
    for (int k2 = 0; k2 < KS; ++k2) {
      float rv = rr[k2];
      ang[0] += rv * wa[k2 * 6 + 0];
      ang[1] += rv * wa[k2 * 6 + 1];
      ang[2] += rv * wa[k2 * 6 + 2];
      ang[3] += rv * wa[k2 * 6 + 3];
    }
  }
  float cs[4], sn[4];
#pragma unroll
  for (int j = 0; j < 4; ++j) { cs[j] = cosf(ang[j]); sn[j] = sinf(ang[j]); }

#define ROT(C, I, J)                                              \
  { float xe = C[I], xo = C[I + 1];                               \
    C[I]     = xe * cs[J] - xo * sn[J];                           \
    C[I + 1] = xe * sn[J] + xo * cs[J]; }

  if (half == 0) {
    ROT(cq, 4, 0) ROT(cq, 6, 1)
    ROT(ck, 4, 0) ROT(ck, 6, 1)
  } else {
    ROT(cq, 0, 0) ROT(cq, 2, 1) ROT(cq, 4, 2) ROT(cq, 6, 3)
    ROT(ck, 0, 0) ROT(ck, 2, 1) ROT(ck, 4, 2) ROT(ck, 6, 3)
  }
#undef ROT

  // store: 8 consecutive floats per lane -> 2x b128 stores
  size_t ob = (size_t)node * 128 + h * 16 + half * 8;
  float4 s;
  s.x=cq[0]; s.y=cq[1]; s.z=cq[2]; s.w=cq[3]; ((float4*)(Qws + ob))[0] = s;
  s.x=cq[4]; s.y=cq[5]; s.z=cq[6]; s.w=cq[7]; ((float4*)(Qws + ob))[1] = s;
  s.x=ck[0]; s.y=ck[1]; s.z=ck[2]; s.w=ck[3]; ((float4*)(Kws + ob))[0] = s;
  s.x=ck[4]; s.y=ck[5]; s.z=ck[6]; s.w=ck[7]; ((float4*)(Kws + ob))[1] = s;
  s.x=cv[0]; s.y=cv[1]; s.z=cv[2]; s.w=cv[3]; ((float4*)(Vws + ob))[0] = s;
  s.x=cv[4]; s.y=cv[5]; s.z=cv[6]; s.w=cv[7]; ((float4*)(Vws + ob))[1] = s;
}

// ---------------- kernel 2: per-edge GEMM + logits + wE ----------------
// grid = EDGE_TILES blocks x 256 threads; wave h handles head h.
__global__ void grit_edge_kernel(const float* __restrict__ ea,
                                 const int* __restrict__ ei,
                                 const _Float16* __restrict__ Ewh,
                                 const _Float16* __restrict__ wewh,
                                 const float* __restrict__ Eb,
                                 const float* __restrict__ web,
                                 const float* __restrict__ Qws, const float* __restrict__ Kws,
                                 float* __restrict__ den, float* __restrict__ exws,
                                 float* __restrict__ wE) {
  __shared__ _Float16 et[16 * 128];  // edge_attr tile, f16
  const int e0 = blockIdx.x * 16;
  const int t  = threadIdx.x;

  {
    int idx = t * 8;
    int row = idx >> 7;
    int colc = idx & 127;
    const float4* p = (const float4*)(ea + (size_t)(e0 + row) * INF + colc);
    float4 a = p[0], b = p[1];
    *(v8h*)&et[idx] = cvt8(a, b);
    if (e0 + 16 < NEDGE)  // prefetch next tile (speculative, dropped if invalid)
      __builtin_prefetch(ea + (size_t)(e0 + 16 + row) * INF + colc, 0, 0);
  }
  __syncthreads();

  const int lane = t & 31;
  const int h    = t >> 5;
  const int col  = lane & 15;       // edge within tile
  const int half = lane >> 4;
  const int e    = e0 + col;
  const int src  = ei[e];
  const int dst  = ei[NEDGE + e];

  v8f c0 = {0,0,0,0,0,0,0,0};   // E_proj[..., :D]  (Ew rows h*32 .. +15)
  v8f c1 = {0,0,0,0,0,0,0,0};   // E_proj[..., D:]  (Ew rows h*32+16 .. +31)
  v8f cb = {0,0,0,0,0,0,0,0};   // e_bias matrix    (wew rows; only m<8 valid)

  const _Float16* a0p = Ewh  + (size_t)(h * 32 + col) * INF;
  const _Float16* a1p = Ewh  + (size_t)(h * 32 + 16 + col) * INF;
  const _Float16* awp = wewh + (size_t)(col < 8 ? col : 7) * INF;  // clamp OOB rows

#pragma unroll 1
  for (int kk = 0; kk < 4; ++kk) {
    v16h B = loadB_lds(&et[col * 128 + kk * 32 + half * 8]);
    int ka = kk * 32 + half * 8;
    c0 = wmma_f16(loadA_f16row(a0p, ka), B, c0);
    c1 = wmma_f16(loadA_f16row(a1p, ka), B, c1);
    cb = wmma_f16(loadA_f16row(awp, ka), B, cb);
  }

  // gather Q_dst/K_src slices: this lane covers dims d = half*8 + 0..7 of head h
  size_t qb = (size_t)dst * 128 + h * 16 + half * 8;
  size_t kb = (size_t)src * 128 + h * 16 + half * 8;
  float4 q0 = ((const float4*)(Qws + qb))[0], q1 = ((const float4*)(Qws + qb))[1];
  float4 k0 = ((const float4*)(Kws + kb))[0], k1 = ((const float4*)(Kws + kb))[1];

  const float SSEM = 0.5f;                   // 1/sqrt(D_SEM=4)
  const float SSTR = 0.28867513459481287f;   // 1/sqrt(D_STR=12)
  float w0 = half ? SSTR : SSEM;  // dims 0..3 semantic only on half 0
  float part = (q0.x*k0.x + q0.y*k0.y + q0.z*k0.z + q0.w*k0.w) * w0
             + (q1.x*k1.x + q1.y*k1.y + q1.z*k1.z + q1.w*k1.w) * SSTR;
  part += __shfl_xor(part, 16, 32);  // combine dim halves across lane pairs

  if (half == 0) {
    float eb = vsel8(cb, h) + web[h];
    float logit = part + eb;
    logit = fminf(CLAMPV, fmaxf(-CLAMPV, logit));
    // logits clamped to +-5 => skip segment-max shift (attn is shift-invariant)
    float ex = expf(logit);
    exws[(size_t)e * HH + h] = ex;
    atomicAdd(&den[(size_t)dst * HH + h], ex);
  }

  // wE: e_t = signed_sqrt((K_src+Q_dst)*E_proj[:,:D]) + E_proj[:,D:]
  float qv[8] = {q0.x,q0.y,q0.z,q0.w,q1.x,q1.y,q1.z,q1.w};
  float kv[8] = {k0.x,k0.y,k0.z,k0.w,k1.x,k1.y,k1.z,k1.w};
  float ot[8];
#pragma unroll
  for (int r = 0; r < 8; ++r) {
    int m = h * 32 + half * 8 + r;
    float mult = c0[r] + Eb[m];
    float addv = c1[r] + Eb[m + 16];
    float v = (qv[r] + kv[r]) * mult;
    float ssq = copysignf(sqrtf(fabsf(v)), v);
    ot[r] = ssq + addv;
  }
  float* op = wE + (size_t)e * 128 + h * 16 + half * 8;
  float4 s;
  s.x=ot[0]; s.y=ot[1]; s.z=ot[2]; s.w=ot[3]; ((float4*)op)[0] = s;
  s.x=ot[4]; s.y=ot[5]; s.z=ot[6]; s.w=ot[7]; ((float4*)op)[1] = s;
}

// ---------------- kernel 3: attn = ex/den, scatter wV ----------------
__global__ void grit_scatter_kernel(const int* __restrict__ ei,
                                    const float* __restrict__ Vws,
                                    const float* __restrict__ exws,
                                    const float* __restrict__ den,
                                    float* __restrict__ wV) {
  size_t gid = (size_t)blockIdx.x * blockDim.x + threadIdx.x;  // NE*H threads
  int e = (int)(gid >> 3);
  int h = (int)(gid & 7);
  int src = ei[e];
  int dst = ei[NEDGE + e];
  float a = exws[(size_t)e * HH + h] / (den[(size_t)dst * HH + h] + 1e-16f);
  const float* vp = Vws + (size_t)src * 128 + h * 16;
  float*       op = wV  + (size_t)dst * 128 + h * 16;
#pragma unroll
  for (int c = 0; c < 4; ++c) {
    float4 v = ((const float4*)vp)[c];
    atomicAdd(op + c * 4 + 0, v.x * a);
    atomicAdd(op + c * 4 + 1, v.y * a);
    atomicAdd(op + c * 4 + 2, v.z * a);
    atomicAdd(op + c * 4 + 3, v.w * a);
  }
}

// ---------------- host launcher ----------------
extern "C" void kernel_launch(void* const* d_in, const int* in_sizes, int n_in,
                              void* d_out, int out_size, void* d_ws, size_t ws_size,
                              hipStream_t stream) {
  (void)in_sizes; (void)n_in; (void)out_size; (void)ws_size;
  const float* x     = (const float*)d_in[0];
  const int*   ei    = (const int*)  d_in[1];
  const float* rrwp  = (const float*)d_in[2];
  const float* ea    = (const float*)d_in[3];
  const float* Qw    = (const float*)d_in[4];
  const float* Qb    = (const float*)d_in[5];
  const float* Kw    = (const float*)d_in[6];
  const float* Kb    = (const float*)d_in[7];
  const float* Vw    = (const float*)d_in[8];
  const float* Vb    = (const float*)d_in[9];
  const float* Ew    = (const float*)d_in[10];
  const float* Eb    = (const float*)d_in[11];
  const float* wew   = (const float*)d_in[12];
  const float* web   = (const float*)d_in[13];
  const float* Wang  = (const float*)d_in[14];

  float* wV = (float*)d_out;                       // (N, H*D)
  float* wE = (float*)d_out + (size_t)NN * 128;    // (NE, H*D)

  // workspace layout (~94.6 MB)
  float* Qws  = (float*)d_ws;                          // N*128 f32
  float* Kws  = Qws + (size_t)NN * 128;                // N*128 f32
  float* Vws  = Kws + (size_t)NN * 128;                // N*128 f32
  float* den  = Vws + (size_t)NN * 128;                // N*H   f32
  float* exws = den + (size_t)NN * HH;                 // NE*H  f32
  // pre-converted f16 weights (one-shot conversion per launch)
  _Float16* Qh   = (_Float16*)(exws + (size_t)NEDGE * HH);  // 128*128
  _Float16* Kh   = Qh + 128 * 128;                          // 128*128
  _Float16* Vh   = Kh + 128 * 128;                          // 128*128
  _Float16* Ewh  = Vh + 128 * 128;                          // 256*128
  _Float16* wewh = Ewh + 256 * 128;                         // 8*128

  grit_zero_kernel<<<(NN * 128 + 255) / 256, 256, 0, stream>>>(wV, den);

  grit_cvt_kernel<<<(128 * 128 + 255) / 256, 256, 0, stream>>>(Qw, Qh, 128 * 128);
  grit_cvt_kernel<<<(128 * 128 + 255) / 256, 256, 0, stream>>>(Kw, Kh, 128 * 128);
  grit_cvt_kernel<<<(128 * 128 + 255) / 256, 256, 0, stream>>>(Vw, Vh, 128 * 128);
  grit_cvt_kernel<<<(256 * 128 + 255) / 256, 256, 0, stream>>>(Ew, Ewh, 256 * 128);
  grit_cvt_kernel<<<(8 * 128 + 255) / 256, 256, 0, stream>>>(wew, wewh, 8 * 128);

  grit_node_kernel<<<NODE_TILES, 256, 0, stream>>>(
      x, rrwp, Qh, Kh, Vh, Qb, Kb, Vb, Wang, Qws, Kws, Vws);

  grit_edge_kernel<<<EDGE_TILES, 256, 0, stream>>>(
      ea, ei, Ewh, wewh, Eb, web, Qws, Kws, den, exws, wE);

  grit_scatter_kernel<<<(NEDGE * HH) / 256, 256, 0, stream>>>(
      ei, Vws, exws, den, wV);
}
